// ApproachNet_1941325218392
// MI455X (gfx1250) — matrix operand order repaired and checked by reading, not compiled, not run on hardware
//
#include <hip/hip_runtime.h>
#include <hip/hip_bf16.h>
#include <math.h>

// ---------------------------------------------------------------------------
// Problem constants (from reference setup_inputs / model definition)
// ---------------------------------------------------------------------------
#define BB    2          // batch
#define NN    2048       // points per cloud
#define KK    64         // neighbors in radius query
#define SS    1000       // grasp samples per batch
#define GFD_  1024       // global feature dim
#define BN    (BB * NN)          // 4096
#define BNK   (BB * NN * KK)     // 262144
#define BS    (BB * SS)          // 2000
#define RU32(x) (((x) + 31) & ~31)

typedef __attribute__((ext_vector_type(16))) _Float16 v16h;
typedef __attribute__((ext_vector_type(8)))  _Float16 v8h;
typedef __attribute__((ext_vector_type(4)))  _Float16 v4h;
typedef __attribute__((ext_vector_type(8)))  float    v8f;
typedef __attribute__((ext_vector_type(4)))  float    v4f;

static inline int ceil_div(int a, int b) { return (a + b - 1) / b; }

// ---------------------------------------------------------------------------
// Weight pre-pass: W[Kd,Nd] f32 row-major  ->  Wt[ldN,ldK] f16, transposed,
// zero-padded.  Makes each lane's B fragment a contiguous 32B run.
// ---------------------------------------------------------------------------
__global__ void convert_wt(const float* __restrict__ W, _Float16* __restrict__ Wt,
                           int Kd, int Nd, int ldK, int total)
{
    int t = blockIdx.x * blockDim.x + threadIdx.x;
    if (t >= total) return;
    int k = t % ldK, n = t / ldK;
    float v = (k < Kd && n < Nd) ? W[(size_t)k * Nd + n] : 0.0f;
    Wt[t] = (_Float16)v;
}

// ---------------------------------------------------------------------------
// WMMA GEMM: Y[M,ldY] = act(X[M,ldK] @ Wt^T + bias), f16 in / f32 accumulate.
// One wave per 16x16 tile. ldK is a multiple of 32 and all pads are zero, so
// the inner loop is branch-free: 4x b128 A loads + cvt, 2x b128 B loads, wmma.
// A/B k order here: contiguous, element e <-> k = 16*hl + e (wave32). act: 0 = identity, 1 = ReLU.
// M must be a multiple of 16 (true for all call sites). Columns n in [Nd,ldY)
// are written as zeros (keeps downstream pads clean).
// ---------------------------------------------------------------------------
__global__ __launch_bounds__(32)
void gemm_wmma_f16(const float* __restrict__ X, const _Float16* __restrict__ Wt,
                   const float* __restrict__ bias, float* __restrict__ Y,
                   int M, int ldK, int Nd, int ldY, int act)
{
    const int lane = threadIdx.x;      // 0..31
    const int hl   = lane >> 4;        // half-wave select
    const int l16  = lane & 15;
    const int M0   = blockIdx.x * 16;
    const int N0   = blockIdx.y * 16;

    const int mA = M0 + l16;                                   // A row (always < M)
    const int n  = N0 + l16;                                   // B column / output col
    const float*    xrow = X  + (size_t)mA * ldK + 16 * hl;    // lane's first K-run
    const _Float16* wrow = Wt + (size_t)n  * ldK + 16 * hl;    // lane's B K-run

    v8f acc = {};
    for (int K0 = 0; K0 < ldK; K0 += 32) {
        if (K0 + 32 < ldK) {                                   // -> global_prefetch_b8
            __builtin_prefetch(xrow + K0 + 32, 0, 0);
            __builtin_prefetch(wrow + K0 + 32, 0, 0);
        }
        // A fragment: element e <-> K = K0 + 16*hl + e (contiguous)
        const v4f* px = (const v4f*)(xrow + K0);
        v4h h0 = __builtin_convertvector(px[0], v4h);
        v4h h1 = __builtin_convertvector(px[1], v4h);
        v4h h2 = __builtin_convertvector(px[2], v4h);
        v4h h3 = __builtin_convertvector(px[3], v4h);
        v8h a01 = __builtin_shufflevector(h0, h1, 0, 1, 2, 3, 4, 5, 6, 7);
        v8h a23 = __builtin_shufflevector(h2, h3, 0, 1, 2, 3, 4, 5, 6, 7);
        v16h a  = __builtin_shufflevector(a01, a23, 0, 1, 2, 3, 4, 5, 6, 7,
                                          8, 9, 10, 11, 12, 13, 14, 15);
        // B fragment: element e <-> K = K0 + 16*hl + e (contiguous f16 in Wt)
        const v8h* pw = (const v8h*)(wrow + K0);
        v8h b0 = pw[0], b1 = pw[1];
        v16h bf = __builtin_shufflevector(b0, b1, 0, 1, 2, 3, 4, 5, 6, 7,
                                          8, 9, 10, 11, 12, 13, 14, 15);
        acc = __builtin_amdgcn_wmma_f32_16x16x32_f16(
            false, a, false, bf, (short)0, acc, false, false);
    }

    // Epilogue: C/D layout -> M = r + 8*hl, N = l16
    const bool nOK = (n < Nd);
    const float bv = nOK ? bias[n] : 0.0f;
#pragma unroll
    for (int r = 0; r < 8; ++r) {
        int m = M0 + r + 8 * hl;
        float v = nOK ? (acc[r] + bv) : 0.0f;
        if (act == 1) v = fmaxf(v, 0.0f);
        Y[(size_t)m * ldY + n] = v;
    }
}

static void gemm(const float* X, const _Float16* Wt, const float* b, float* Y,
                 int M, int ldK, int Nd, int ldY, int act, hipStream_t s)
{
    dim3 g(M / 16, ldY / 16);
    gemm_wmma_f16<<<g, dim3(32), 0, s>>>(X, Wt, b, Y, M, ldK, Nd, ldY, act);
}

// ---------------------------------------------------------------------------
// K=64 nearest-neighbor selection per query (ascending distance; first 3
// entries double as the k=3 interpolation neighbors).
// ---------------------------------------------------------------------------
__global__ __launch_bounds__(256)
void knn_kernel(const float* __restrict__ pos, int* __restrict__ knn_idx,
                float* __restrict__ knn_d2)
{
    const int q = blockIdx.x;                  // 0..B*N-1
    const int b = q / NN, i = q % NN;
    const int tid = threadIdx.x;
    __shared__ float d2s[NN];
    __shared__ float rv[256];
    __shared__ int   ri[256];

    const float px = pos[((size_t)b * NN + i) * 3 + 0];
    const float py = pos[((size_t)b * NN + i) * 3 + 1];
    const float pz = pos[((size_t)b * NN + i) * 3 + 2];
    for (int j = tid; j < NN; j += 256) {
        float dx = pos[((size_t)b * NN + j) * 3 + 0] - px;
        float dy = pos[((size_t)b * NN + j) * 3 + 1] - py;
        float dz = pos[((size_t)b * NN + j) * 3 + 2] - pz;
        d2s[j] = dx * dx + dy * dy + dz * dz;
    }
    __syncthreads();

    for (int k = 0; k < KK; ++k) {
        float best = 3.4e38f; int bi = -1;
        for (int j = tid; j < NN; j += 256)
            if (d2s[j] < best) { best = d2s[j]; bi = j; }
        rv[tid] = best; ri[tid] = bi;
        __syncthreads();
        for (int s = 128; s > 0; s >>= 1) {
            if (tid < s && rv[tid + s] < rv[tid]) { rv[tid] = rv[tid + s]; ri[tid] = ri[tid + s]; }
            __syncthreads();
        }
        if (tid == 0) {
            knn_idx[(size_t)q * KK + k] = ri[0];
            knn_d2 [(size_t)q * KK + k] = rv[0];
            d2s[ri[0]] = 3.4e38f;
        }
        __syncthreads();
    }
}

// ---------------------------------------------------------------------------
// Gather neighbor features into padded rows:
//   Xg[row, 0:C)   = feat[b, j]
//   Xg[row, C:C+3) = pos[b,j] - pos[b,i]
//   Xg[row, C+3:ld) = 0
// ---------------------------------------------------------------------------
__global__ void gather_sa(const float* __restrict__ feat, int C,
                          const float* __restrict__ pos,
                          const int* __restrict__ knn,
                          float* __restrict__ Xg, int ld, int total)
{
    int t = blockIdx.x * blockDim.x + threadIdx.x;
    if (t >= total) return;
    int col = t % ld;
    int row = t / ld;                      // (b*N + i)*K + k
    int bi  = row / KK;                    // b*N + i
    int b   = bi / NN;
    float v = 0.0f;
    if (col < C + 3) {
        int j = knn[row];
        if (col < C) v = feat[((size_t)b * NN + j) * C + col];
        else {
            int d = col - C;
            v = pos[((size_t)b * NN + j) * 3 + d] - pos[(size_t)bi * 3 + d];
        }
    }
    Xg[t] = v;
}

// Masked max-aggregation over K neighbors (radius mask, -1e30 fill)
__global__ void max_agg(const float* __restrict__ H, const float* __restrict__ d2,
                        float r2, float* __restrict__ Y, int Hc, int total)
{
    int t = blockIdx.x * blockDim.x + threadIdx.x;
    if (t >= total) return;
    int col = t % Hc;
    int row = t / Hc;                       // b*N + i
    float best = -1e30f;
    for (int k = 0; k < KK; ++k) {
        if (d2[(size_t)row * KK + k] <= r2)
            best = fmaxf(best, H[((size_t)row * KK + k) * Hc + col]);
    }
    Y[t] = best;
}

// Row-wise concat [A | B | zero-pad] into ld-wide rows
__global__ void concat2(const float* __restrict__ A, int Ca,
                        const float* __restrict__ Bm, int Cb,
                        float* __restrict__ Y, int ld, int total)
{
    int t = blockIdx.x * blockDim.x + threadIdx.x;
    if (t >= total) return;
    int col = t % ld, row = t / ld;
    float v = 0.0f;
    if (col < Ca)           v = A[(size_t)row * Ca + col];
    else if (col < Ca + Cb) v = Bm[(size_t)row * Cb + (col - Ca)];
    Y[t] = v;
}

// Concat [broadcast g[b] | X[row] | pad]
__global__ void concat_gx(const float* __restrict__ g, int Cg,
                          const float* __restrict__ X, int Cx,
                          float* __restrict__ Y, int ld, int total)
{
    int t = blockIdx.x * blockDim.x + threadIdx.x;
    if (t >= total) return;
    int col = t % ld, row = t / ld, b = row / NN;
    float v = 0.0f;
    if (col < Cg)           v = g[(size_t)b * Cg + col];
    else if (col < Cg + Cx) v = X[(size_t)row * Cx + (col - Cg)];
    Y[t] = v;
}

// Global max over points: g[b,f] = max_i X[b*N+i, f]   (X stride = C)
__global__ void rowmax_g(const float* __restrict__ X, float* __restrict__ g, int C, int total)
{
    int t = blockIdx.x * blockDim.x + threadIdx.x;
    if (t >= total) return;
    int b = t / C, c = t % C;
    float best = -3.4e38f;
    for (int i = 0; i < NN; ++i)
        best = fmaxf(best, X[((size_t)b * NN + i) * C + c]);
    g[t] = best;
}

// knn_interpolate k=3 (first 3 ascending-distance neighbors), stride C in/out
__global__ void knn_interp3(const float* __restrict__ X, const int* __restrict__ knn,
                            const float* __restrict__ d2, float* __restrict__ Y,
                            int C, int total)
{
    int t = blockIdx.x * blockDim.x + threadIdx.x;
    if (t >= total) return;
    int col = t % C, row = t / C, b = row / NN;
    float acc = 0.0f, wsum = 0.0f;
    for (int k = 0; k < 3; ++k) {
        float w = 1.0f / fmaxf(d2[(size_t)row * KK + k], 1e-16f);
        int   j = knn[(size_t)row * KK + k];
        acc  += w * X[((size_t)b * NN + j) * C + col];
        wsum += w;
    }
    Y[t] = acc / wsum;
}

__global__ void sigmoid_k(const float* __restrict__ X, int stride,
                          float* __restrict__ Y, int total)
{
    int t = blockIdx.x * blockDim.x + threadIdx.x;
    if (t >= total) return;
    Y[t] = 1.0f / (1.0f + expf(-X[(size_t)t * stride]));
}

// Gumbel top-S multinomial sampling without replacement (per batch)
__global__ __launch_bounds__(256)
void gumbel_topk(const float* __restrict__ approach, const float* __restrict__ noise,
                 int* __restrict__ idxS)
{
    const int b = blockIdx.x, tid = threadIdx.x;
    __shared__ float sc[NN];
    __shared__ float rv[256];
    __shared__ int   ri[256];
    for (int j = tid; j < NN; j += 256) {
        float u = fminf(fmaxf(noise[(size_t)b * NN + j], 1e-9f), 1.0f - 1e-9f);
        float gum = -logf(-logf(u));
        sc[j] = logf(approach[(size_t)b * NN + j] + 1e-12f) + gum;
    }
    __syncthreads();
    for (int s = 0; s < SS; ++s) {
        float best = -3.4e38f; int bi = -1;
        for (int j = tid; j < NN; j += 256)
            if (sc[j] > best) { best = sc[j]; bi = j; }
        rv[tid] = best; ri[tid] = bi;
        __syncthreads();
        for (int r = 128; r > 0; r >>= 1) {
            if (tid < r && rv[tid + r] > rv[tid]) { rv[tid] = rv[tid + r]; ri[tid] = ri[tid + r]; }
            __syncthreads();
        }
        if (tid == 0) { idxS[(size_t)b * SS + s] = ri[0]; sc[ri[0]] = -3.4e38f; }
        __syncthreads();
    }
}

// Gather C columns per sampled index into ld-padded dst (+optional dense dst2)
__global__ void gather_cols(const float* __restrict__ src, int C,
                            const int* __restrict__ idxS,
                            float* __restrict__ dst, int ld,
                            float* __restrict__ dst2, int total)
{
    int t = blockIdx.x * blockDim.x + threadIdx.x;
    if (t >= total) return;
    int col = t % ld, r = t / ld, b = r / SS;
    float v = 0.0f;
    if (col < C) {
        int i = idxS[r];
        v = src[((size_t)b * NN + i) * C + col];
        if (dst2) dst2[(size_t)r * C + col] = v;
    }
    dst[t] = v;
}

// Grasp-net input rows: [x1(128) | x2(256) | g[r%B](1024) | af(64)] = 1472
__global__ void build_feats(const float* __restrict__ x1, const float* __restrict__ x2,
                            const float* __restrict__ g,  const float* __restrict__ af,
                            const int* __restrict__ idxS, float* __restrict__ F, int total)
{
    const int CF = 128 + 256 + GFD_ + 64;   // 1472
    int t = blockIdx.x * blockDim.x + threadIdx.x;
    if (t >= total) return;
    int col = t % CF, r = t / CF, b = r / SS;
    int i = idxS[r];
    float v;
    if (col < 128)        v = x1[((size_t)b * NN + i) * 128 + col];
    else if (col < 384)   v = x2[((size_t)b * NN + i) * 256 + (col - 128)];
    else if (col < 1408)  v = g[(size_t)(r % BB) * GFD_ + (col - 384)];   // tile(g,(S,1))
    else                  v = af[(size_t)r * 64 + (col - 1408)];
    F[t] = v;
}

__global__ void zero_losses(float* gl, float* al)
{
    if (blockIdx.x == 0 && threadIdx.x == 0) { *gl = 0.0f; *al = 0.0f; }
}

__global__ void mse_loss(const float* __restrict__ p, const float* __restrict__ gt,
                         float* __restrict__ out, int total)
{
    int t = blockIdx.x * blockDim.x + threadIdx.x;
    if (t >= total) return;
    float d = p[t] - gt[t];
    atomicAdd(out, d * d / (float)total);
}

__global__ void bce_loss(const float* __restrict__ ap, const float* __restrict__ sc,
                         float* __restrict__ out, int total)
{
    int t = blockIdx.x * blockDim.x + threadIdx.x;
    if (t >= total) return;
    float p = fminf(fmaxf(ap[t], 1e-12f), 1.0f - 1e-12f);
    float a = (sc[t] > 0.0f) ? 1.0f : 0.0f;
    float bce = -(a * logf(p) + (1.0f - a) * logf(1.0f - p));
    atomicAdd(out, bce / (float)total);
}

// ---------------------------------------------------------------------------
// Orchestration
// ---------------------------------------------------------------------------
extern "C" void kernel_launch(void* const* d_in, const int* in_sizes, int n_in,
                              void* d_out, int out_size, void* d_ws, size_t ws_size,
                              hipStream_t stream)
{
    (void)in_sizes; (void)n_in; (void)out_size; (void)ws_size;
    const float* pos    = (const float*)d_in[0];   // [B,N,3]
    const float* y      = (const float*)d_in[1];   // [B,N,16]
    const float* ascore = (const float*)d_in[2];   // [B,N]
    const float* noise  = (const float*)d_in[3];   // [B,N]

    // Params flattened in pytree order (dict keys sorted alphabetically):
#define IN(i) ((const float*)d_in[(i)])
    const float *apW0=IN(4),  *apB0=IN(5),  *apW1=IN(6),  *apB1=IN(7);
    const float *f1W0=IN(8),  *f1B0=IN(9),  *f1W1=IN(10), *f1B1=IN(11), *f1W2=IN(12), *f1B2=IN(13);
    const float *f2W0=IN(14), *f2B0=IN(15), *f2W1=IN(16), *f2B1=IN(17);
    const float *f3W0=IN(18), *f3B0=IN(19), *f3W1=IN(20), *f3B1=IN(21);
    const float *grW0=IN(22), *grB0=IN(23), *grW1=IN(24), *grB1=IN(25), *grW2=IN(26), *grB2=IN(27);
    const float *hdW0=IN(28), *hdB0=IN(29), *hdW1=IN(30), *hdB1=IN(31);
    const float *s1W0=IN(32), *s1B0=IN(33), *s1W1=IN(34), *s1B1=IN(35), *s1W2=IN(36), *s1B2=IN(37);
    const float *s2W0=IN(38), *s2B0=IN(39), *s2W1=IN(40), *s2B1=IN(41), *s2W2=IN(42), *s2B2=IN(43);
    const float *s3W0=IN(44), *s3B0=IN(45), *s3W1=IN(46), *s3B1=IN(47), *s3W2=IN(48), *s3B2=IN(49);
#undef IN

    // Output layout (flat, return order)
    float* outp = (float*)d_out;
    float* out_grasp = outp;                       // 2000*16
    float* out_app   = outp + 32000;               // 4096
    float* out_gt    = outp + 36096;               // 32000
    float* out_gl    = outp + 68096;               // 1
    float* out_al    = outp + 68097;               // 1
    float* out_ap    = outp + 68098;               // 6000

    // Workspace carve (float elements; all block sizes multiples of 4 -> 16B aligned)
    float* base = (float*)d_ws;
    size_t o = 0;
    int*   knn_idx = (int*)(base + o);   o += (size_t)BNK;
    float* knn_d2  = base + o;           o += (size_t)BNK;
    int*   idxS    = (int*)(base + o);   o += (size_t)BS;
    float* Xg      = base + o;           o += (size_t)BNK * 160;   // max padded gather
    float* bufA    = base + o;           o += (size_t)BNK * 256;
    float* bufB    = base + o;           o += (size_t)BNK * 256;
    float* x1      = base + o;           o += (size_t)BN * 128;
    float* x2      = base + o;           o += (size_t)BN * 256;
    float* gfeat   = base + o;           o += (size_t)BB * GFD_;
    float* fpA     = base + o;           o += (size_t)BN * 1280;
    float* fpB     = base + o;           o += (size_t)BN * 1280;
    float* feats   = base + o;           o += (size_t)BS * 1472;
    float* gbufA   = base + o;           o += (size_t)BS * 512;
    float* gbufB   = base + o;           o += (size_t)BS * 512;
    float* ap_ws   = base + o;           o += (size_t)BS * 32;
    float* af      = base + o;           o += (size_t)BS * 64;
    _Float16* wtbase = (_Float16*)(base + o);
    size_t wo = 0;

#define EW(total) <<<ceil_div((total), 256), 256, 0, stream>>>
    // Weight pre-conversion: f32 [Kd,Nd] -> f16 transposed [ldN, RU32(Kd)]
#define CVT(name, Wp, Kd, ldN) \
    _Float16* name = wtbase + wo; \
    { int ldk_ = RU32(Kd); int tot_ = (ldN) * ldk_; \
      convert_wt EW(tot_)(Wp, name, Kd, /*Nd=*/0x7fffffff & 0 ? 0 : 0, ldk_, tot_); }

    // (CVT macro above can't carry Nd cleanly; expand explicitly instead)
#undef CVT
#define CVT(name, Wp, Kd, Nd, ldN) \
    _Float16* name = wtbase + wo; \
    { int ldk_ = RU32(Kd); int tot_ = (ldN) * ldk_; \
      convert_wt EW(tot_)(Wp, name, Kd, Nd, ldk_, tot_); wo += (size_t)tot_; }

    CVT(s1T0, s1W0, 6,    64,   64)   CVT(s1T1, s1W1, 64,  64,   64)
    CVT(s1T2, s1W2, 64,   128,  128)
    CVT(s2T0, s2W0, 131,  128,  128)  CVT(s2T1, s2W1, 128, 128,  128)
    CVT(s2T2, s2W2, 128,  256,  256)
    CVT(s3T0, s3W0, 259,  256,  256)  CVT(s3T1, s3W1, 256, 512,  512)
    CVT(s3T2, s3W2, 512,  1024, 1024)
    CVT(f3T0, f3W0, 1280, 256,  256)  CVT(f3T1, f3W1, 256, 256,  256)
    CVT(f2T0, f2W0, 384,  256,  256)  CVT(f2T1, f2W1, 256, 128,  128)
    CVT(f1T0, f1W0, 131,  128,  128)  CVT(f1T1, f1W1, 128, 128,  128)
    CVT(f1T2, f1W2, 128,  128,  128)
    CVT(hdT0, hdW0, 128,  64,   64)   CVT(hdT1, hdW1, 64,  1,    16)
    CVT(apT0, apW0, 3,    32,   32)   CVT(apT1, apW1, 32,  64,   64)
    CVT(grT0, grW0, 1472, 512,  512)  CVT(grT1, grW1, 512, 256,  256)
    CVT(grT2, grW2, 256,  16,   16)
#undef CVT

    // ---- KNN (shared by sa1/sa2 masks and k=3 interpolation) ----
    knn_kernel<<<BN, 256, 0, stream>>>(pos, knn_idx, knn_d2);

    // ---- SA1: [pos_j, dpos](6, ld32) -> MLP[6,64,64,128] -> masked max ----
    gather_sa EW(BNK * 32)(pos, 3, pos, knn_idx, Xg, 32, BNK * 32);
    gemm(Xg,   s1T0, s1B0, bufA, BNK, 32,  64,  64,  1, stream);
    gemm(bufA, s1T1, s1B1, bufB, BNK, 64,  64,  64,  1, stream);
    gemm(bufB, s1T2, s1B2, bufA, BNK, 64,  128, 128, 0, stream);
    max_agg EW(BN * 128)(bufA, knn_d2, 0.1f * 0.1f, x1, 128, BN * 128);

    // ---- SA2: [x1_j, dpos](131, ld160) -> MLP[131,128,128,256] -> masked max ----
    gather_sa EW(BNK * 160)(x1, 128, pos, knn_idx, Xg, 160, BNK * 160);
    gemm(Xg,   s2T0, s2B0, bufA, BNK, 160, 128, 128, 1, stream);
    gemm(bufA, s2T1, s2B1, bufB, BNK, 128, 128, 128, 1, stream);
    gemm(bufB, s2T2, s2B2, bufA, BNK, 128, 256, 256, 0, stream);
    max_agg EW(BN * 256)(bufA, knn_d2, 0.2f * 0.2f, x2, 256, BN * 256);

    // ---- SA3: [x2, pos](259, ld288) -> MLP[259,256,512,1024] -> max_N -> g ----
    concat2 EW(BN * 288)(x2, 256, pos, 3, fpA, 288, BN * 288);
    gemm(fpA, s3T0, s3B0, fpB, BN, 288, 256,  256,  1, stream);
    gemm(fpB, s3T1, s3B1, fpA, BN, 256, 512,  512,  1, stream);
    gemm(fpA, s3T2, s3B2, fpB, BN, 512, 1024, 1024, 0, stream);
    rowmax_g EW(BB * GFD_)(fpB, gfeat, GFD_, BB * GFD_);

    // ---- FP3: [g bcast, x2](1280) -> MLP[1280,256,256] -> knn_interp(3) ----
    concat_gx EW(BN * 1280)(gfeat, GFD_, x2, 256, fpA, 1280, BN * 1280);
    gemm(fpA, f3T0, f3B0, fpB, BN, 1280, 256, 256, 1, stream);
    gemm(fpB, f3T1, f3B1, fpA, BN, 256,  256, 256, 0, stream);
    knn_interp3 EW(BN * 256)(fpA, knn_idx, knn_d2, fpB, 256, BN * 256);

    // ---- FP2: [h, x1](384) -> MLP[384,256,128] -> knn_interp(3) ----
    concat2 EW(BN * 384)(fpB, 256, x1, 128, fpA, 384, BN * 384);
    gemm(fpA, f2T0, f2B0, fpB, BN, 384, 256, 256, 1, stream);
    gemm(fpB, f2T1, f2B1, fpA, BN, 256, 128, 128, 0, stream);
    knn_interp3 EW(BN * 128)(fpA, knn_idx, knn_d2, fpB, 128, BN * 128);

    // ---- FP1: [h, pos](131, ld160) -> MLP[131,128,128,128] ----
    concat2 EW(BN * 160)(fpB, 128, pos, 3, fpA, 160, BN * 160);
    gemm(fpA, f1T0, f1B0, fpB, BN, 160, 128, 128, 1, stream);
    gemm(fpB, f1T1, f1B1, fpA, BN, 128, 128, 128, 1, stream);
    gemm(fpA, f1T2, f1B2, fpB, BN, 128, 128, 128, 0, stream);

    // ---- Head + sigmoid -> approach ----
    gemm(fpB, hdT0, hdB0, fpA, BN, 128, 64, 64, 1, stream);
    gemm(fpA, hdT1, hdB1, fpB, BN, 64,  1,  16, 0, stream);
    sigmoid_k EW(BN)(fpB, 16, out_app, BN);

    // ---- Gumbel top-S sampling ----
    gumbel_topk<<<BB, 256, 0, stream>>>(out_app, noise, idxS);

    // ---- Gathers: gt (y, dense 16), approach points (pos, ld32 + dense 3) ----
    gather_cols EW(BS * 16)(y,   16, idxS, out_gt, 16, nullptr, BS * 16);
    gather_cols EW(BS * 32)(pos, 3,  idxS, ap_ws,  32, out_ap,  BS * 32);

    // ---- ap_enc MLP[3,32,64] ----
    gemm(ap_ws, apT0, apB0, gbufB, BS, 32, 32, 32, 1, stream);
    gemm(gbufB, apT1, apB1, af,    BS, 32, 64, 64, 0, stream);

    // ---- Grasp head MLP[1472,512,256,16] -> d_out ----
    build_feats EW(BS * 1472)(x1, x2, gfeat, af, idxS, feats, BS * 1472);
    gemm(feats, grT0, grB0, gbufA,     BS, 1472, 512, 512, 1, stream);
    gemm(gbufA, grT1, grB1, gbufB,     BS, 512,  256, 256, 1, stream);
    gemm(gbufB, grT2, grB2, out_grasp, BS, 256,  16,  16,  0, stream);

    // ---- Losses ----
    zero_losses<<<1, 32, 0, stream>>>(out_gl, out_al);
    mse_loss EW(BS * 16)(out_grasp, out_gt, out_gl, BS * 16);
    bce_loss EW(BN)(out_app, ascore, out_al, BN);

#undef EW
}
